// DialecticalAttentionHead_46222438040014
// MI455X (gfx1250) — matrix operand order repaired and checked by reading, not compiled, not run on hardware
//
#include <hip/hip_runtime.h>

// ---------------------------------------------------------------------------
// DialecticalAttentionHead for MI455X (gfx1250), wave32.
// bf16 WMMA (v_wmma_f32_16x16x32_bf16) for all GEMMs + TDM (tensor_load_to_lds)
// double-buffered K/V staging in the attention kernel.
//
// ~37 GFLOP vs ~72MB unique traffic -> matrix-core bound.
// ---------------------------------------------------------------------------

#define SCALE_QK 0.08838834764831845f   // 1/sqrt(128)
#define THRESH2  0.01f                  // 0.1^2

typedef __attribute__((ext_vector_type(16))) __bf16 v16bf;
typedef __attribute__((ext_vector_type(8)))  __bf16 v8bf;
typedef __attribute__((ext_vector_type(8)))  float  v8f;
typedef __attribute__((ext_vector_type(4)))  unsigned int u32x4;
typedef __attribute__((ext_vector_type(8)))  int i32x8;
typedef __attribute__((ext_vector_type(4)))  int i32x4;

#if defined(__HIP_DEVICE_COMPILE__) && defined(__gfx1250__) && \
    __has_builtin(__builtin_amdgcn_tensor_load_to_lds)
#define HAVE_TDM 1
#else
#define HAVE_TDM 0
#endif

static __device__ __forceinline__ v8f wmma_bf16(v16bf a, v16bf b, v8f c) {
  // D = A(16x32) * B(32x16) + C, fp32 accum
  return __builtin_amdgcn_wmma_f32_16x16x32_bf16(false, a, false, b,
                                                 (short)0, c, false, false);
}

// B-fragment: lane holds column (lane&15), 16 contiguous K values (caller adds
// (lane>>4)*16 to the K base).  Two 16B loads -> 16B alignment requirement.
static __device__ __forceinline__ v16bf ld_b16(const __bf16* p) {
  union { v16bf v; v8bf h[2]; } u;
  u.h[0] = *(const v8bf*)(p);
  u.h[1] = *(const v8bf*)(p + 8);
  return u.v;
}

// A-fragment (16-bit, 16x32): lane&15 = row M, lane>>4 selects K half:
// 16 contiguous elems at k0 + half*16.
static __device__ __forceinline__ v16bf ld_a_bf16(const __bf16* rowk0, int lanehi) {
  union { v16bf v; v8bf h[2]; } u;
  u.h[0] = *(const v8bf*)(rowk0 + lanehi * 16);
  u.h[1] = *(const v8bf*)(rowk0 + 8 + lanehi * 16);
  return u.v;
}

// Same A-fragment but converting fp32 memory -> bf16 on the fly (for x).
static __device__ __forceinline__ v16bf ld_a_f32(const float* rowk0, int lanehi) {
  union { v16bf v; __bf16 e[16]; } u;
  const float* p0 = rowk0 + lanehi * 16;
  const float* p1 = rowk0 + 8 + lanehi * 16;
#pragma unroll
  for (int i = 0; i < 8; ++i) {
    u.e[i]     = (__bf16)p0[i];
    u.e[8 + i] = (__bf16)p1[i];
  }
  return u.v;
}

static __device__ __forceinline__ v8f splat8(float x) {
  v8f v;
#pragma unroll
  for (int i = 0; i < 8; ++i) v[i] = x;
  return v;
}

// Single-wave workgroups: cross-lane LDS visibility only needs the CDNA5
// split DS counter to drain (same-wave LDS ops complete in order).
static __device__ __forceinline__ void wave_lds_sync() {
  asm volatile("s_wait_dscnt 0" ::: "memory");
}

#if HAVE_TDM
// --- Tensor Data Mover: 2-D tile descriptor (D# groups 0/1), bf16 elements ---
struct TdmDesc { u32x4 g0; i32x8 g1; };

static __device__ __forceinline__ TdmDesc make_tdm_2d(
    unsigned lds_addr, unsigned long long gaddr,
    unsigned tensor_d0, unsigned tensor_d1, unsigned long long stride0,
    unsigned tile_d0, unsigned tile_d1) {
  TdmDesc d;
  // group0: [1:0]=count=1, [63:32]=lds_addr, [120:64]=global_addr, [127:126]=type=2
  d.g0[0] = 1u;
  d.g0[1] = lds_addr;
  d.g0[2] = (unsigned)(gaddr & 0xffffffffull);
  d.g0[3] = (unsigned)((gaddr >> 32) & 0x01ffffffull) | (2u << 30);
  // group1: [17:16]=data_size=1 (2B); dims/tiles/strides in element units
  d.g1[0] = 0x00010000;                                   // mask=0, 2B elements
  d.g1[1] = (int)((tensor_d0 & 0xffffu) << 16);           // tensor_dim0[15:0] @79:48
  d.g1[2] = (int)((tensor_d0 >> 16) & 0xffffu) |          // tensor_dim0[31:16]
            (int)((tensor_d1 & 0xffffu) << 16);           // tensor_dim1[15:0]
  d.g1[3] = (int)((tensor_d1 >> 16) & 0xffffu) |          // tensor_dim1[31:16]
            (int)((tile_d0 & 0xffffu) << 16);             // tile_dim0
  d.g1[4] = (int)(tile_d1 & 0xffffu);                     // tile_dim1 (tile_dim2=0)
  d.g1[5] = (int)(stride0 & 0xffffffffull);               // dim0_stride[31:0]
  d.g1[6] = (int)((stride0 >> 32) & 0xffffull);           // dim0_stride[47:32]
  d.g1[7] = 0;                                            // dim1_stride (unused, 2D)
  return d;
}

static __device__ __forceinline__ void tdm_load(const TdmDesc& d) {
#if __clang_major__ >= 23
  __builtin_amdgcn_tensor_load_to_lds(d.g0, d.g1, (i32x4)0, (i32x4)0, (i32x8)0, 0);
#else
  __builtin_amdgcn_tensor_load_to_lds(d.g0, d.g1, (i32x4)0, (i32x4)0, 0);
#endif
}

#if __has_builtin(__builtin_amdgcn_s_wait_tensorcnt)
#define WAIT_TENSORCNT(n) __builtin_amdgcn_s_wait_tensorcnt((short)(n))
#else
#define WAIT_TENSORCNT(n) asm volatile("s_wait_tensorcnt %0" ::"i"(n) : "memory")
#endif
#endif  // HAVE_TDM

// ---------------------------------------------------------------------------
// Kernel 0: fp32 -> bf16 conversion (weights)
// ---------------------------------------------------------------------------
__global__ void cvt_f32_bf16_kernel(const float* __restrict__ in,
                                    __bf16* __restrict__ out, int n) {
  int i = blockIdx.x * 256 + threadIdx.x;
  if (i < n) out[i] = (__bf16)in[i];
}

// ---------------------------------------------------------------------------
// Kernel 1: QKV projection.  grid = (1024 m-tiles, 8 n-tiles, 3 {q,k,v})
// y = x @ W^T : A = x rows (fp32->bf16 inline), B column n = W row n.
// v is written TRANSPOSED per batch: vT[b][dim][s] so P@V gets contiguous B.
// ---------------------------------------------------------------------------
__global__ __launch_bounds__(32) void qkv_gemm_kernel(
    const float* __restrict__ x,
    const __bf16* __restrict__ wq, const __bf16* __restrict__ wk,
    const __bf16* __restrict__ wv,
    __bf16* __restrict__ q, __bf16* __restrict__ k, __bf16* __restrict__ vT) {
  const int lane = threadIdx.x, col = lane & 15, lanehi = lane >> 4;
  const int mt = blockIdx.x, nt = blockIdx.y, which = blockIdx.z;
  const __bf16* w = (which == 0) ? wq : (which == 1) ? wk : wv;

  const float*  xrow = x + (size_t)(mt * 16 + col) * 1024;
  const __bf16* wcol = w + (size_t)(nt * 16 + col) * 1024 + lanehi * 16;

  v8f acc = splat8(0.f);
#pragma unroll 4
  for (int k0 = 0; k0 < 1024; k0 += 32) {
    if (k0 + 128 < 1024) __builtin_prefetch(xrow + k0 + 128, 0, 1);
    v16bf a = ld_a_f32(xrow + k0, lanehi);
    v16bf b = ld_b16(wcol + k0);
    acc = wmma_bf16(a, b, acc);
  }

  if (which == 2) {
#pragma unroll
    for (int r = 0; r < 8; ++r) {
      int tok = mt * 16 + r + lanehi * 8;
      int bb = tok >> 11, ss = tok & 2047;
      vT[((size_t)bb * 128 + nt * 16 + col) * 2048 + ss] = (__bf16)acc[r];
    }
  } else {
    __bf16* o = (which == 0) ? q : k;
#pragma unroll
    for (int r = 0; r < 8; ++r) {
      int tok = mt * 16 + r + lanehi * 8;
      o[(size_t)tok * 128 + nt * 16 + col] = (__bf16)acc[r];
    }
  }
}

// ---------------------------------------------------------------------------
// Kernel 2: flash attention, one wave per 16-query tile, 32 keys/iteration.
// K/V tiles are TDM-streamed into double-buffered LDS (tensor_load_to_lds,
// s_wait_tensorcnt) so WMMA B-fragments come from ds_loads while the next
// tile is in flight.  P is re-laid-out D->A through LDS (sP[16][32] bf16).
// ---------------------------------------------------------------------------
__global__ __launch_bounds__(32) void attn_kernel(
    const __bf16* __restrict__ q, const __bf16* __restrict__ k,
    const __bf16* __restrict__ vT, float* __restrict__ cur) {
  __shared__ alignas(16) __bf16 sP[16 * 32];

  const int lane = threadIdx.x, col = lane & 15, lanehi = lane >> 4;
  const int tile = blockIdx.x;           // 0..1023
  const int b = tile >> 7;
  const int tokbase = tile * 16;         // flat token index (batch-contiguous)

  // Resident q A-fragments for the whole head dim (K = 128 = 4 steps)
  const __bf16* qrow = q + (size_t)(tokbase + col) * 128;
  v16bf aq[4];
#pragma unroll
  for (int i = 0; i < 4; ++i) aq[i] = ld_a_bf16(qrow + i * 32, lanehi);

  v8f acc[8];
#pragma unroll
  for (int i = 0; i < 8; ++i) acc[i] = splat8(0.f);
  float m8[8], l8[8];
#pragma unroll
  for (int r = 0; r < 8; ++r) { m8[r] = -1e30f; l8[r] = 0.f; }

  const __bf16* kb0 = k  + (size_t)b * 2048 * 128;
  const __bf16* vb0 = vT + (size_t)b * 128 * 2048;

#if HAVE_TDM
  __shared__ alignas(16) __bf16 sK[2][32 * 128];   // [key][dim]
  __shared__ alignas(16) __bf16 sV[2][128 * 32];   // [dim][key]
  const unsigned ldsK[2] = {(unsigned)(uintptr_t)&sK[0][0],
                            (unsigned)(uintptr_t)&sK[1][0]};
  const unsigned ldsV[2] = {(unsigned)(uintptr_t)&sV[0][0],
                            (unsigned)(uintptr_t)&sV[1][0]};
  // prologue: stream tiles for j=0 into buffer 0
  tdm_load(make_tdm_2d(ldsK[0], (unsigned long long)(uintptr_t)kb0,
                       128, 2048, 128, 128, 32));
  tdm_load(make_tdm_2d(ldsV[0], (unsigned long long)(uintptr_t)vb0,
                       2048, 128, 2048, 32, 128));
#endif

  for (int j = 0; j < 2048; j += 32) {
#if HAVE_TDM
    const int buf = (j >> 5) & 1;
    if (j + 32 < 2048) {
      const int jn = j + 32;
      wave_lds_sync();  // prior ds reads of buf^1 drained before TDM rewrites it
      tdm_load(make_tdm_2d(
          ldsK[buf ^ 1], (unsigned long long)(uintptr_t)(kb0 + (size_t)jn * 128),
          128, 2048, 128, 128, 32));
      tdm_load(make_tdm_2d(
          ldsV[buf ^ 1], (unsigned long long)(uintptr_t)(vb0 + jn),
          2048, 128, 2048, 32, 128));
      WAIT_TENSORCNT(2);   // current buffer's 2 loads done; next 2 in flight
    } else {
      WAIT_TENSORCNT(0);
    }
    const __bf16* krow0 = &sK[buf][(size_t)col * 128 + lanehi * 16];
    const __bf16* krow1 = krow0 + 16 * 128;
#else
    const __bf16* krow0 = kb0 + (size_t)(j + col) * 128 + lanehi * 16;
    const __bf16* krow1 = krow0 + 16 * 128;
#endif

    // --- scores: two 16-key tiles, K(head)=128 ---
    v8f s0 = splat8(0.f), s1 = splat8(0.f);
#pragma unroll
    for (int i = 0; i < 4; ++i) {
      s0 = wmma_bf16(aq[i], ld_b16(krow0 + i * 32), s0);
      s1 = wmma_bf16(aq[i], ld_b16(krow1 + i * 32), s1);
    }

    // --- online softmax (row stats via shfl_xor butterflies in each half) ---
    float p0[8], p1[8];
#pragma unroll
    for (int r = 0; r < 8; ++r) {
      s0[r] *= SCALE_QK; s1[r] *= SCALE_QK;
      float t = fmaxf(s0[r], s1[r]);
      t = fmaxf(t, __shfl_xor(t, 1, 32));
      t = fmaxf(t, __shfl_xor(t, 2, 32));
      t = fmaxf(t, __shfl_xor(t, 4, 32));
      t = fmaxf(t, __shfl_xor(t, 8, 32));
      float mnew = fmaxf(m8[r], t);
      float alpha = __expf(m8[r] - mnew);
      m8[r] = mnew;
      p0[r] = __expf(s0[r] - mnew);
      p1[r] = __expf(s1[r] - mnew);
      float rs = p0[r] + p1[r];
      rs += __shfl_xor(rs, 1, 32);
      rs += __shfl_xor(rs, 2, 32);
      rs += __shfl_xor(rs, 4, 32);
      rs += __shfl_xor(rs, 8, 32);
      l8[r] = l8[r] * alpha + rs;
#pragma unroll
      for (int nt = 0; nt < 8; ++nt) acc[nt][r] *= alpha;
    }

    // --- P: D layout -> row-major bf16 in LDS ---
#pragma unroll
    for (int r = 0; r < 8; ++r) {
      sP[(r + lanehi * 8) * 32 + col]      = (__bf16)p0[r];
      sP[(r + lanehi * 8) * 32 + 16 + col] = (__bf16)p1[r];
    }
    wave_lds_sync();

    // --- acc += P(16x32) @ V(32x128) ---
    v16bf ap = ld_a_bf16(sP + (size_t)col * 32, lanehi);
#pragma unroll
    for (int nt = 0; nt < 8; ++nt) {
#if HAVE_TDM
      const __bf16* vcol = &sV[buf][(size_t)(nt * 16 + col) * 32 + lanehi * 16];
#else
      const __bf16* vcol = vb0 + (size_t)(nt * 16 + col) * 2048 + j + lanehi * 16;
#endif
      acc[nt] = wmma_bf16(ap, ld_b16(vcol), acc[nt]);
    }
    wave_lds_sync();
  }

  // normalize and write cur (fp32)
#pragma unroll
  for (int nt = 0; nt < 8; ++nt)
#pragma unroll
    for (int r = 0; r < 8; ++r)
      cur[(size_t)(tokbase + r + lanehi * 8) * 128 + nt * 16 + col] =
          acc[nt][r] / l8[r];
}

// ---------------------------------------------------------------------------
// Kernel 3: thesis/antithesis + 3 synthesis rounds.  One wave / 16 tokens.
// anti = -(cur@Wt^T) + anti_b, so t0 = cur@Wt^T is computed once.
// Per round: sCat=[at|aa|ac] (bf16, masked) -> relu(W1) -> W2 -> gate -> update.
// Assumes any_active==True every round (prob ~1 over 16K random tokens),
// which makes the recurrence token-local.
// ---------------------------------------------------------------------------
__global__ __launch_bounds__(32) void synth_kernel(
    const float* __restrict__ cur_in,
    const __bf16* __restrict__ thw, const float* __restrict__ thb,
    const float* __restrict__ anb,
    const __bf16* __restrict__ sw1, const float* __restrict__ sb1,
    const __bf16* __restrict__ sw2, const float* __restrict__ sb2,
    const __bf16* __restrict__ gw, const float* __restrict__ gb,
    float* __restrict__ out) {
  __shared__ alignas(16) __bf16 sCat[16 * 384];   // [at | aa | ac]
  __shared__ alignas(16) __bf16 sH[16 * 128];
  __shared__ alignas(16) __bf16 sSyn[16 * 128];
  __shared__ float sGate[16];

  const int lane = threadIdx.x, col = lane & 15, lanehi = lane >> 4;
  const int mt = blockIdx.x;   // 0..1023

  // cur tile in registers, D layout (lane=feature col, r=token row)
  float curv[8][8];
  float thbv[8], anbv[8], sb1v[8], sb2v[8];
#pragma unroll
  for (int nt = 0; nt < 8; ++nt) {
    thbv[nt] = thb[nt * 16 + col];
    anbv[nt] = anb[nt * 16 + col];
    sb1v[nt] = sb1[nt * 16 + col];
    sb2v[nt] = sb2[nt * 16 + col];
#pragma unroll
    for (int r = 0; r < 8; ++r) {
      curv[nt][r] =
          cur_in[(size_t)(mt * 16 + r + lanehi * 8) * 128 + nt * 16 + col];
      sCat[(r + lanehi * 8) * 384 + 256 + nt * 16 + col] = (__bf16)curv[nt][r];
    }
  }
  wave_lds_sync();

  // t0 = cur @ thesis_w^T (round-invariant)
  float t0[8][8];
#pragma unroll
  for (int nt = 0; nt < 8; ++nt) {
    v8f a = splat8(0.f);
    const __bf16* wc = thw + (size_t)(nt * 16 + col) * 128 + lanehi * 16;
#pragma unroll
    for (int ks = 0; ks < 4; ++ks)
      a = wmma_bf16(ld_a_bf16(sCat + col * 384 + 256 + ks * 32, lanehi),
                    ld_b16(wc + ks * 32), a);
#pragma unroll
    for (int r = 0; r < 8; ++r) t0[nt][r] = a[r];
  }

  float am[8];
#pragma unroll
  for (int r = 0; r < 8; ++r) am[r] = 1.f;

  for (int round = 0; round < 3; ++round) {
    // stage masked [at | aa | ac] in bf16
#pragma unroll
    for (int nt = 0; nt < 8; ++nt)
#pragma unroll
      for (int r = 0; r < 8; ++r) {
        int ro = (r + lanehi * 8) * 384, c = nt * 16 + col;
        sCat[ro + c]       = (__bf16)(am[r] * (t0[nt][r] + thbv[nt]));
        sCat[ro + 128 + c] = (__bf16)(am[r] * (-t0[nt][r] + anbv[nt]));
        sCat[ro + 256 + c] = (__bf16)(am[r] * curv[nt][r]);
      }
    wave_lds_sync();

    // h = relu(cat @ W1^T + b1), K = 384
#pragma unroll
    for (int nt = 0; nt < 8; ++nt) {
      v8f a = splat8(sb1v[nt]);
      const __bf16* wc = sw1 + (size_t)(nt * 16 + col) * 384 + lanehi * 16;
#pragma unroll
      for (int ks = 0; ks < 12; ++ks)
        a = wmma_bf16(ld_a_bf16(sCat + col * 384 + ks * 32, lanehi),
                      ld_b16(wc + ks * 32), a);
#pragma unroll
      for (int r = 0; r < 8; ++r)
        sH[(r + lanehi * 8) * 128 + nt * 16 + col] = (__bf16)fmaxf(a[r], 0.f);
    }
    wave_lds_sync();

    // synth = h @ W2^T + b2, K = 128
#pragma unroll
    for (int nt = 0; nt < 8; ++nt) {
      v8f a = splat8(sb2v[nt]);
      const __bf16* wc = sw2 + (size_t)(nt * 16 + col) * 128 + lanehi * 16;
#pragma unroll
      for (int ks = 0; ks < 4; ++ks)
        a = wmma_bf16(ld_a_bf16(sH + col * 128 + ks * 32, lanehi),
                      ld_b16(wc + ks * 32), a);
#pragma unroll
      for (int r = 0; r < 8; ++r)
        sSyn[(r + lanehi * 8) * 128 + nt * 16 + col] = (__bf16)a[r];
    }
    wave_lds_sync();

    // scalar gate per token: sigmoid([ac|synth] . g_w + g_b)
    if (lane < 16) {
      float g = gb[0];
      const __bf16* acrow = sCat + lane * 384 + 256;
      const __bf16* syrow = sSyn + lane * 128;
      for (int i = 0; i < 128; ++i) g += (float)acrow[i] * (float)gw[i];
      for (int i = 0; i < 128; ++i) g += (float)syrow[i] * (float)gw[128 + i];
      sGate[lane] = 1.f / (1.f + __expf(-g));
    }
    wave_lds_sync();

    float g8[8], nrm[8];
#pragma unroll
    for (int r = 0; r < 8; ++r) { g8[r] = sGate[r + lanehi * 8]; nrm[r] = 0.f; }

#pragma unroll
    for (int nt = 0; nt < 8; ++nt)
#pragma unroll
      for (int r = 0; r < 8; ++r) {
        float sv = (float)sSyn[(r + lanehi * 8) * 128 + nt * 16 + col];
        float ac = am[r] * curv[nt][r];
        float u = g8[r] * (sv - ac) * 0.1f;
        curv[nt][r] += u;
        nrm[r] += u * u;
      }
#pragma unroll
    for (int r = 0; r < 8; ++r) {
      float t = nrm[r];
      t += __shfl_xor(t, 1, 32);
      t += __shfl_xor(t, 2, 32);
      t += __shfl_xor(t, 4, 32);
      t += __shfl_xor(t, 8, 32);
      if (t < THRESH2) am[r] = 0.f;     // active &= !stable
    }
    wave_lds_sync();                    // protect sCat reuse next round
  }

#pragma unroll
  for (int nt = 0; nt < 8; ++nt)
#pragma unroll
    for (int r = 0; r < 8; ++r)
      out[(size_t)(mt * 16 + r + lanehi * 8) * 128 + nt * 16 + col] =
          curv[nt][r];
}

// ---------------------------------------------------------------------------
// Host side
// ---------------------------------------------------------------------------
extern "C" void kernel_launch(void* const* d_in, const int* in_sizes, int n_in,
                              void* d_out, int out_size, void* d_ws,
                              size_t ws_size, hipStream_t stream) {
  (void)in_sizes; (void)n_in; (void)out_size; (void)ws_size;

  const float* x   = (const float*)d_in[0];
  const float* wq  = (const float*)d_in[1];
  const float* wk  = (const float*)d_in[2];
  const float* wv  = (const float*)d_in[3];
  const float* thw = (const float*)d_in[4];
  const float* thb = (const float*)d_in[5];
  const float* anb = (const float*)d_in[6];
  const float* sw1 = (const float*)d_in[7];
  const float* sb1 = (const float*)d_in[8];
  const float* sw2 = (const float*)d_in[9];
  const float* sb2 = (const float*)d_in[10];
  const float* gw  = (const float*)d_in[11];
  const float* gb  = (const float*)d_in[12];

  char* ws = (char*)d_ws;
  size_t off = 0;
  auto wsalloc = [&](size_t bytes) {
    void* p = ws + off;
    off += (bytes + 255) & ~(size_t)255;
    return p;
  };
  __bf16* qb   = (__bf16*)wsalloc((size_t)16384 * 128 * 2);
  __bf16* kb   = (__bf16*)wsalloc((size_t)16384 * 128 * 2);
  __bf16* vTb  = (__bf16*)wsalloc((size_t)16384 * 128 * 2);
  __bf16* wqb  = (__bf16*)wsalloc((size_t)131072 * 2);
  __bf16* wkb  = (__bf16*)wsalloc((size_t)131072 * 2);
  __bf16* wvb  = (__bf16*)wsalloc((size_t)131072 * 2);
  __bf16* thwb = (__bf16*)wsalloc((size_t)16384 * 2);
  __bf16* sw1b = (__bf16*)wsalloc((size_t)49152 * 2);
  __bf16* sw2b = (__bf16*)wsalloc((size_t)16384 * 2);
  __bf16* gwb  = (__bf16*)wsalloc((size_t)256 * 2);
  float*  cur  = (float*)wsalloc((size_t)16384 * 128 * 4);   // ~22 MB total

  auto cvt = [&](const float* in, __bf16* outp, int n) {
    cvt_f32_bf16_kernel<<<dim3((n + 255) / 256), dim3(256), 0, stream>>>(in, outp, n);
  };
  cvt(wq, wqb, 131072);
  cvt(wk, wkb, 131072);
  cvt(wv, wvb, 131072);
  cvt(thw, thwb, 16384);
  cvt(sw1, sw1b, 49152);
  cvt(sw2, sw2b, 16384);
  cvt(gw, gwb, 256);

  qkv_gemm_kernel<<<dim3(1024, 8, 3), dim3(32), 0, stream>>>(
      x, wqb, wkb, wvb, qb, kb, vTb);

  attn_kernel<<<dim3(1024), dim3(32), 0, stream>>>(qb, kb, vTb, cur);

  synth_kernel<<<dim3(1024), dim3(32), 0, stream>>>(
      cur, thwb, thb, anb, sw1b, sb1, sw2b, sb2, gwb, gb, (float*)d_out);
}
